// VC_MODEL_36069135352130
// MI455X (gfx1250) — matrix operand order repaired and checked by reading, not compiled, not run on hardware
//
#include <hip/hip_runtime.h>
#include <math.h>

typedef __attribute__((ext_vector_type(16))) _Float16 v16h;
typedef __attribute__((ext_vector_type(8)))  _Float16 v8h;
typedef __attribute__((ext_vector_type(4)))  _Float16 v4h;
typedef __attribute__((ext_vector_type(8)))  float    v8f;
typedef __attribute__((ext_vector_type(4)))  float    v4f;
typedef v4f uv4f __attribute__((aligned(4)));   // 4-byte-aligned float4 (unaligned-capable load)

union ABfrag { v16h v; v8h h8[2]; };

// ---------------- small helpers ----------------

__device__ inline float blockReduceSum(float v, float* sbuf) {
    __syncthreads();
    const int tid = threadIdx.x;
    #pragma unroll
    for (int off = 16; off > 0; off >>= 1) v += __shfl_xor(v, off, 32);
    const int wv = tid >> 5;
    if ((tid & 31) == 0) sbuf[wv] = v;
    __syncthreads();
    const int nw = blockDim.x >> 5;
    v = (tid < nw) ? sbuf[tid] : 0.f;
    if (tid < 32) {
        #pragma unroll
        for (int off = 16; off > 0; off >>= 1) v += __shfl_xor(v, off, 32);
    }
    if (tid == 0) sbuf[0] = v;
    __syncthreads();
    return sbuf[0];
}

// ---------------- kernel 0: weight conversion ----------------
// w1 [128][240][4] f32 -> w1h [128][960] f16  (K = ci*4+kw, flat order identical)
// w2 [120][128][3] f32 -> w2h [128][384] f16  (K = kw*128+ci, kw-major; rows >=120 zero)

__global__ __launch_bounds__(256) void prep_kernel(
    const float* __restrict__ w1, const float* __restrict__ w2,
    _Float16* __restrict__ w1h, _Float16* __restrict__ w2h) {
    int idx = blockIdx.x * 256 + threadIdx.x;
    if (idx < 122880) {
        w1h[idx] = (_Float16)w1[idx];
    } else if (idx < 122880 + 49152) {
        int j  = idx - 122880;
        int co = j / 384;
        int r  = j - co * 384;
        int kw = r >> 7, ci = r & 127;
        w2h[j] = (co < 120) ? (_Float16)w2[(co * 128 + ci) * 3 + kw] : (_Float16)0.f;
    }
}

// ---------------- kernel 0b: normalized codebook ----------------

__global__ __launch_bounds__(128) void en_kernel(
    const float* __restrict__ embed, float* __restrict__ En) {
    __shared__ float sb[32];
    const int k = blockIdx.x;      // 0..31
    const int c = threadIdx.x;     // 0..127
    float e = (c < 120) ? embed[k * 120 + c] : 0.f;
    float ss = blockReduceSum(e * e, sb);
    float nrm = sqrtf(ss);
    if (c < 120) En[k * 120 + c] = e / nrm;
}

// ---------------- kernel 1: conv1 (stride2 pad1 K4) + ReLU, WMMA f16 ----------------
// out h1 [16][128][4096] f16.  K = ci*4+kw (960), input pos = 2t-1+kw.

__global__ __launch_bounds__(256) void conv1_kernel(
    const float* __restrict__ x, const float* __restrict__ b1,
    const _Float16* __restrict__ w1h, _Float16* __restrict__ h1) {
    __shared__ _Float16 Bt[64 * 32] __attribute__((aligned(32)));  // [n][k] (K transposed)
    const int t0  = blockIdx.x * 64;
    const int b   = blockIdx.y;
    const int tid = threadIdx.x;
    const int wv  = tid >> 5, lane = tid & 31;
    const int m0  = wv * 16;
    const int row = lane & 15, hi = lane >> 4;

    v8f acc[4];
    #pragma unroll
    for (int i = 0; i < 4; ++i)
        #pragma unroll
        for (int j = 0; j < 8; ++j) acc[i][j] = 0.f;

    for (int k0 = 0; k0 < 960; k0 += 32) {
        __syncthreads();
        // Fill: 512 groups of (ci_rel 0..7, n 0..63); each group = 4 consecutive
        // input floats (kw=0..3) -> one 8-byte DS store of 4 f16.
        #pragma unroll
        for (int i = 0; i < 2; ++i) {
            int g = tid + i * 256;
            int ci_rel = g >> 6, n = g & 63;
            int ci = (k0 >> 2) + ci_rel;
            int p0 = 2 * (t0 + n) - 1;
            const float* xp = x + ((b * 240 + ci) << 13);
            float f[4];
            if (p0 >= 0 && p0 + 3 < 8192) {
                uv4f q = *(const uv4f*)(xp + p0);
                f[0] = q.x; f[1] = q.y; f[2] = q.z; f[3] = q.w;
            } else {
                #pragma unroll
                for (int j = 0; j < 4; ++j) {
                    int p  = p0 + j;
                    int pc = p < 0 ? 0 : (p > 8191 ? 8191 : p);
                    float t = xp[pc];
                    f[j] = ((unsigned)p < 8192u) ? t : 0.f;
                }
            }
            v4h hh;
            #pragma unroll
            for (int j = 0; j < 4; ++j) hh[j] = (_Float16)f[j];
            *(v4h*)(Bt + n * 32 + ci_rel * 4) = hh;
        }
        __syncthreads();
        // A fragment: lane row = M, halves = K {hi*16 + 0..15}
        ABfrag a;
        const _Float16* ap = w1h + (m0 + row) * 960 + k0 + hi * 16;
        a.h8[0] = *(const v8h*)ap;
        a.h8[1] = *(const v8h*)(ap + 8);
        #pragma unroll
        for (int nt = 0; nt < 4; ++nt) {
            // B fragment: lane col = N, halves = K {hi*16 + 0..15}
            const v16h bf = *(const v16h*)(Bt + (nt * 16 + row) * 32 + hi * 16);
            acc[nt] = __builtin_amdgcn_wmma_f32_16x16x32_f16(
                false, a.v, false, bf, (short)0, acc[nt], false, false);
        }
    }
    // D layout: lane: N = lane&15, VGPR r: M = r + hi*8
    #pragma unroll
    for (int r = 0; r < 8; ++r) {
        int co = m0 + hi * 8 + r;
        float bv = b1[co];
        #pragma unroll
        for (int nt = 0; nt < 4; ++nt) {
            float v = acc[nt][r] + bv;
            v = v > 0.f ? v : 0.f;
            h1[((b * 128 + co) << 12) + t0 + nt * 16 + row] = (_Float16)v;
        }
    }
}

// ---------------- kernel 2: conv2 (stride1 pad1 K3), WMMA f16 ----------------
// out h [16][120][4096] f32.  K = kw*128+ci (384, kw-major), input pos = t-1+kw.
// Each 32-wide K chunk has fixed kw and 32 consecutive ci -> coalesced f16 loads.

__global__ __launch_bounds__(256) void conv2_kernel(
    const _Float16* __restrict__ h1, const float* __restrict__ b2,
    const _Float16* __restrict__ w2h, float* __restrict__ h) {
    __shared__ _Float16 Bt[64 * 32] __attribute__((aligned(32)));
    const int t0  = blockIdx.x * 64;
    const int b   = blockIdx.y;
    const int tid = threadIdx.x;
    const int wv  = tid >> 5, lane = tid & 31;
    const int m0  = wv * 16;
    const int row = lane & 15, hi = lane >> 4;

    v8f acc[4];
    #pragma unroll
    for (int i = 0; i < 4; ++i)
        #pragma unroll
        for (int j = 0; j < 8; ++j) acc[i][j] = 0.f;

    for (int k0 = 0; k0 < 384; k0 += 32) {
        const int kw  = k0 >> 7;     // uniform within chunk
        const int ci0 = k0 & 127;
        __syncthreads();
        #pragma unroll
        for (int i = 0; i < 8; ++i) {
            int idx  = tid + i * 256;
            int kloc = idx >> 6, n = idx & 63;
            int ci = ci0 + kloc;
            int p  = t0 + n - 1 + kw;
            int pc = p < 0 ? 0 : (p > 4095 ? 4095 : p);
            _Float16 v = h1[((b * 128 + ci) << 12) + pc];
            if ((unsigned)p >= 4096u) v = (_Float16)0.f;
            Bt[n * 32 + kloc] = v;
        }
        __syncthreads();
        ABfrag a;
        const _Float16* ap = w2h + (m0 + row) * 384 + k0 + hi * 16;
        a.h8[0] = *(const v8h*)ap;
        a.h8[1] = *(const v8h*)(ap + 8);
        #pragma unroll
        for (int nt = 0; nt < 4; ++nt) {
            const v16h bf = *(const v16h*)(Bt + (nt * 16 + row) * 32 + hi * 16);
            acc[nt] = __builtin_amdgcn_wmma_f32_16x16x32_f16(
                false, a.v, false, bf, (short)0, acc[nt], false, false);
        }
    }
    #pragma unroll
    for (int r = 0; r < 8; ++r) {
        int d = m0 + hi * 8 + r;
        if (d < 120) {
            float bv = b2[d];
            #pragma unroll
            for (int nt = 0; nt < 4; ++nt)
                h[((b * 120 + d) << 12) + t0 + nt * 16 + row] = acc[nt][r] + bv;
        }
    }
}

// ---------------- kernel 3: per-(b,c) time stats ----------------

__global__ __launch_bounds__(256) void stats_kernel(
    const float* __restrict__ h, float* __restrict__ meanb,
    float* __restrict__ stdb, float* __restrict__ outstd) {
    __shared__ float sb[32];
    const int rc = blockIdx.x;  // b*120+c
    const float* p = h + (size_t)rc * 4096;
    float s = 0.f;
    for (int i = threadIdx.x; i < 4096; i += 256) s += p[i];
    s = blockReduceSum(s, sb);
    const float mean = s * (1.f / 4096.f);
    float ss = 0.f;
    for (int i = threadIdx.x; i < 4096; i += 256) {
        float d = p[i] - mean;
        ss += d * d;
    }
    ss = blockReduceSum(ss, sb);
    if (threadIdx.x == 0) {
        float st = sqrtf(ss) + 1e-4f;
        meanb[rc]  = mean;
        stdb[rc]   = st;
        outstd[rc] = st;   // std output [B,120,1]
    }
}

// ---------------- kernel 4: channel norm + VQ + q_after + diff partials ----------------

__global__ __launch_bounds__(256) void vq_kernel(
    const float* __restrict__ h, const float* __restrict__ meanb,
    const float* __restrict__ stdb, const float* __restrict__ En,
    const float* __restrict__ embed, float* __restrict__ qout,
    float* __restrict__ diffpart) {
    __shared__ float hs[120 * 65];    // [c][t] padded pitch
    __shared__ float zb[8 * 128];     // per-wave z
    __shared__ float Ens[32 * 121];   // padded pitch
    __shared__ float mns[120];
    __shared__ float sds[120];
    __shared__ float warr[8];

    const int b   = blockIdx.y;
    const int t0  = blockIdx.x * 64;
    const int tid = threadIdx.x;
    const int wv  = tid >> 5, lane = tid & 31;

    for (int idx = tid; idx < 120 * 64; idx += 256) {
        int r = idx >> 6, col = idx & 63;
        hs[r * 65 + col] = h[((b * 120 + r) << 12) + t0 + col];
    }
    for (int idx = tid; idx < 32 * 120; idx += 256) {
        int k = idx / 120, c = idx - k * 120;
        Ens[k * 121 + c] = En[idx];
    }
    if (tid < 120) {
        mns[tid] = meanb[b * 120 + tid];
        sds[tid] = stdb[b * 120 + tid];
    }
    __syncthreads();

    float dsum = 0.f;
    for (int it = 0; it < 8; ++it) {
        const int tl = it * 8 + wv;  // this wave's local t
        float xn[4];
        float n2 = 0.f;
        #pragma unroll
        for (int m = 0; m < 4; ++m) {
            int c = lane + 32 * m;
            float u = 0.f;
            if (c < 120) {
                float v = hs[c * 65 + tl];
                u = (v - mns[c]) / sds[c];
            }
            xn[m] = u;
            n2 += u * u;
        }
        #pragma unroll
        for (int off = 16; off > 0; off >>= 1) n2 += __shfl_xor(n2, off, 32);
        const float rn = 1.f / sqrtf(n2);
        #pragma unroll
        for (int m = 0; m < 4; ++m) {
            int c = lane + 32 * m;
            if (c < 120) zb[wv * 128 + c] = xn[m] * rn;
        }
        // lane == code index k; argmin(dist) == argmax(z . En[k]) since ||En||=||z||=1
        float dot = 0.f;
        const float* zp = zb + wv * 128;
        const float* ep = Ens + lane * 121;
        for (int c = 0; c < 120; ++c) dot += zp[c] * ep[c];
        float bd = dot;
        int   bk = lane;
        #pragma unroll
        for (int off = 16; off > 0; off >>= 1) {
            float od = __shfl_xor(bd, off, 32);
            int   ok = __shfl_xor(bk, off, 32);
            if (od > bd || (od == bd && ok < bk)) { bd = od; bk = ok; }
        }
        const float* eb = embed + bk * 120;  // lookup uses UNnormalized table
        #pragma unroll
        for (int m = 0; m < 4; ++m) {
            int c = lane + 32 * m;
            if (c < 120) {
                float q = eb[c];
                float z = xn[m] * rn;
                float d = q - z;
                dsum += d * d;
                hs[c * 65 + tl] = q;   // reuse tile for coalesced q_after writeout
            }
        }
    }
    #pragma unroll
    for (int off = 16; off > 0; off >>= 1) dsum += __shfl_xor(dsum, off, 32);
    if (lane == 0) warr[wv] = dsum;
    __syncthreads();
    if (tid == 0) {
        float s = 0.f;
        #pragma unroll
        for (int i = 0; i < 8; ++i) s += warr[i];
        diffpart[b * 64 + blockIdx.x] = s;
    }
    for (int idx = tid; idx < 120 * 64; idx += 256) {
        int r = idx >> 6, col = idx & 63;
        qout[((b * 120 + r) << 12) + t0 + col] = hs[r * 65 + col];
    }
}

// ---------------- kernel 5: speaker raw stats ----------------

__global__ __launch_bounds__(256) void sp_kernel(
    const float* __restrict__ h, const float* __restrict__ q,
    float* __restrict__ spraw) {
    __shared__ float sb[32];
    const int rc = blockIdx.x;
    const float* hp = h + (size_t)rc * 4096;
    const float* qp = q + (size_t)rc * 4096;
    float s = 0.f;
    for (int i = threadIdx.x; i < 4096; i += 256) s += hp[i] - qp[i];
    s = blockReduceSum(s, sb);
    if (threadIdx.x == 0) spraw[rc] = s * (1.f / 4096.f);
}

// ---------------- kernel 6: sp normalization + diff finalize ----------------

__global__ __launch_bounds__(128) void final_kernel(
    const float* __restrict__ spraw, const float* __restrict__ diffpart,
    float* __restrict__ outsp, float* __restrict__ outdiff) {
    __shared__ float sb[32];
    const int b = blockIdx.x;
    const int c = threadIdx.x;
    float v = (c < 120) ? spraw[b * 120 + c] : 0.f;
    float ss = blockReduceSum(v * v, sb);
    float nrm = sqrtf(ss) + 1e-4f;
    if (c < 120) outsp[b * 120 + c] = v / nrm * (1.f / 3.f);
    if (b == 0 && c == 0) {
        float s = 0.f;
        for (int i = 0; i < 1024; ++i) s += diffpart[i];  // fixed order: deterministic
        outdiff[0] = s * (1.f / 7864320.f);               // mean over 16*4096*120
    }
}

// ---------------- launcher ----------------

extern "C" void kernel_launch(void* const* d_in, const int* in_sizes, int n_in,
                              void* d_out, int out_size, void* d_ws, size_t ws_size,
                              hipStream_t stream) {
    (void)in_sizes; (void)n_in; (void)out_size; (void)ws_size;
    const float* x     = (const float*)d_in[0];  // [16,240,8192]
    const float* w1    = (const float*)d_in[1];  // [128,240,4]
    const float* b1    = (const float*)d_in[2];  // [128]
    const float* w2    = (const float*)d_in[3];  // [120,128,3]
    const float* b2    = (const float*)d_in[4];  // [120]
    const float* embed = (const float*)d_in[5];  // [32,120]

    char* ws = (char*)d_ws;
    _Float16* w1h   = (_Float16*)(ws);              // 245760 B
    _Float16* w2h   = (_Float16*)(ws + 245760);     //  98304 B
    _Float16* h1    = (_Float16*)(ws + 344064);     // 16777216 B  [16,128,4096] f16
    float*    h     = (float*)(ws + 17121280);      // 31457280 B  [16,120,4096] f32
    float*    meanb = (float*)(ws + 48578560);      //   7680 B
    float*    stdb  = (float*)(ws + 48586240);      //   7680 B
    float*    En    = (float*)(ws + 48593920);      //  15360 B
    float*    spraw = (float*)(ws + 48609280);      //   7680 B
    float*    dpart = (float*)(ws + 48616960);      //   4096 B  -> total ~46.4 MB

    float* qout    = (float*)d_out;       // [16,120,4096]
    float* outsp   = qout + 7864320;      // [16,120]
    float* outstd  = outsp + 1920;        // [16,120]
    float* outdiff = outstd + 1920;       // [1]

    prep_kernel <<<(122880 + 49152 + 255) / 256, 256, 0, stream>>>(w1, w2, w1h, w2h);
    en_kernel   <<<32, 128, 0, stream>>>(embed, En);
    conv1_kernel<<<dim3(64, 16), 256, 0, stream>>>(x, b1, w1h, h1);
    conv2_kernel<<<dim3(64, 16), 256, 0, stream>>>(h1, b2, w2h, h);
    stats_kernel<<<1920, 256, 0, stream>>>(h, meanb, stdb, outstd);
    vq_kernel   <<<dim3(64, 16), 256, 0, stream>>>(h, meanb, stdb, En, embed, qout, dpart);
    sp_kernel   <<<1920, 256, 0, stream>>>(h, qout, spraw);
    final_kernel<<<16, 128, 0, stream>>>(spraw, dpart, outsp, outdiff);
}
